// FusionSubNet_51299089383489
// MI455X (gfx1250) — matrix operand order repaired and checked by reading, not compiled, not run on hardware
//
#include <hip/hip_runtime.h>
#include <hip/hip_bf16.h>
#include <math.h>

#define LSEQ 4096
#define IN   512
#define IN2  1024
#define G3   1536
#define NB   8
#define GRID_H 0.4f

typedef __attribute__((ext_vector_type(16))) __bf16 v16bf;
typedef __attribute__((ext_vector_type(8)))  __bf16 v8bf;
typedef __attribute__((ext_vector_type(8)))  float  v8f;
typedef __attribute__((ext_vector_type(4)))  int    v4i;

#if defined(__gfx1250__) && __has_builtin(__builtin_amdgcn_global_load_async_to_lds_b128)
#define HAVE_ASYNC_LDS 1
#else
#define HAVE_ASYNC_LDS 0
#endif

union V16 { v16bf v; v8bf h[2]; };

__device__ __forceinline__ __bf16 f2bf(float f) {
  union { float f; unsigned u; } v; v.f = f;
  unsigned r = v.u + 0x7FFFu + ((v.u >> 16) & 1u);   // round-to-nearest-even
  unsigned short hs = (unsigned short)(r >> 16);
  __bf16 o; __builtin_memcpy(&o, &hs, 2); return o;
}

__device__ __forceinline__ float sigmoidf_d(float x) { return 1.f / (1.f + __expf(-x)); }

// 16-byte global -> LDS stage; async copy (ASYNCcnt-tracked) on gfx1250.
__device__ __forceinline__ void lds_stage_16B(const __bf16* gp, __bf16* lp) {
#if HAVE_ASYNC_LDS
  __builtin_amdgcn_global_load_async_to_lds_b128(
      (__attribute__((address_space(1))) v4i*)gp,
      (__attribute__((address_space(3))) v4i*)lp, 0, 0);
#else
  *(float4*)lp = *(const float4*)gp;
#endif
}

__device__ __forceinline__ void wait_async_lds() {
#if HAVE_ASYNC_LDS
#if __has_builtin(__builtin_amdgcn_s_wait_asynccnt)
  __builtin_amdgcn_s_wait_asynccnt(0);
#else
  asm volatile("s_wait_asynccnt 0" ::: "memory");
#endif
#endif
}

// ---------------------------------------------------------------------------
// C(M,N) = A(M,K) @ W(N,K)^T, bf16 in / f32 accum via v_wmma_f32_16x16x32_bf16.
// Block = 256 threads = 8 waves in a 2(M) x 4(N) grid; block tile 64M x 256N.
// A panel (64 x 32 bf16) is staged into LDS with double-buffered async copies
// and shared by the 4 N-wave columns; each wave computes a 32M x 64N strip
// (2 A-frags x 4 B-frags -> 8 WMMAs per K-step).
// mode 0: += bias[col]; mode 1: relu; mode 2: sigmoid.
// Grid must be exactly (M/64)*(N/256) blocks.
// ---------------------------------------------------------------------------
__global__ __launch_bounds__(256)
void wmma_gemm(const __bf16* __restrict__ A, const __bf16* __restrict__ W,
               const float* __restrict__ bias, float* __restrict__ C,
               int M, int N, int K, int mode) {
  const int tid  = threadIdx.x;
  const int lane = tid & 31;
  const int wv   = tid >> 5;           // 0..7
  const int wm   = wv >> 2;            // 0..1 : wave's 32-row half of block tile
  const int wn   = wv & 3;             // 0..3 : wave's 64-col strip

  const int nBlk = N >> 8;             // N / 256
  const int m0 = (blockIdx.x / nBlk) << 6;
  const int n0 = ((blockIdx.x % nBlk) << 8) + (wn << 6);

  __shared__ __align__(16) __bf16 Abuf[2][64 * 32];

  // cooperative staging map: thread -> (row, 16B chunk) of the 64x32 A panel
  const int srow = tid >> 2;           // 0..63
  const int sq   = (tid & 3) << 3;     // bf16 offset 0,8,16,24

  // ISA 16-bit A (16x32): lane<16 -> M=lane, K pairs 0..7 & 16..23;
  //                        lane>=16 -> M=lane-16, K pairs 8..15 & 24..31.
  const int kHalfA = (lane >> 4) << 3;  // 0 or 8
  // ISA 16-bit B (32x16): lane<16 -> N=lane, K 0..7 & 16..23; lane>=16 -> K 8..15 & 24..31.
  const int kHalfB = (lane >> 4) << 3;  // 0 or 8
  const int nLane  = lane & 15;
  const int mLane  = lane & 15;

  v8f acc[2][4];
  #pragma unroll
  for (int mt = 0; mt < 2; ++mt)
    #pragma unroll
    for (int s = 0; s < 4; ++s)
      #pragma unroll
      for (int r = 0; r < 8; ++r) acc[mt][s][r] = 0.f;

  const int nk = K >> 5;
  const __bf16* Ag = A + (size_t)(m0 + srow) * K + sq;

  // prologue: stage K-step 0 into buffer 0
  lds_stage_16B(Ag, &Abuf[0][srow * 32 + sq]);

  for (int it = 0; it < nk; ++it) {
    wait_async_lds();
    __syncthreads();                         // A panel for step `it` visible
    if (it + 1 < nk)                         // stream next panel while computing
      lds_stage_16B(Ag + ((it + 1) << 5), &Abuf[(it + 1) & 1][srow * 32 + sq]);

    const __bf16* Ab = &Abuf[it & 1][0];
    V16 afr[2];
    #pragma unroll
    for (int mt = 0; mt < 2; ++mt) {
      const __bf16* ap = Ab + (size_t)((wm << 5) + (mt << 4) + mLane) * 32 + kHalfA;
      afr[mt].h[0] = *(const v8bf*)ap;        // K = kHalfA + 0..7
      afr[mt].h[1] = *(const v8bf*)(ap + 16); // K = 16 + kHalfA + 0..7
    }

    const int k0 = it << 5;
    #pragma unroll
    for (int s = 0; s < 4; ++s) {
      const __bf16* Bp = W + (size_t)(n0 + (s << 4) + nLane) * K + k0 + kHalfB;
      V16 bfr;
      bfr.h[0] = *(const v8bf*)Bp;
      bfr.h[1] = *(const v8bf*)(Bp + 16);
      __builtin_prefetch(Bp + 32, 0, 3);      // speculative; dropped past end
      #pragma unroll
      for (int mt = 0; mt < 2; ++mt)
        acc[mt][s] = __builtin_amdgcn_wmma_f32_16x16x32_bf16(
            false, afr[mt].v, false, bfr.v, (short)0, acc[mt][s], false, false);
    }
    __syncthreads();                         // all reads done before re-staging
  }

  // C/D layout: VGPR r -> row +r (lanes<16) or +8+r (lanes>=16)
  const int rAdd = (lane >> 4) << 3;
  #pragma unroll
  for (int mt = 0; mt < 2; ++mt) {
    #pragma unroll
    for (int s = 0; s < 4; ++s) {
      const int col = n0 + (s << 4) + nLane;
      const float bv = (mode == 0) ? bias[col] : 0.f;
      #pragma unroll
      for (int r = 0; r < 8; ++r) {
        float v = acc[mt][s][r] + bv;
        if (mode == 1)      v = v > 0.f ? v : 0.f;
        else if (mode == 2) v = sigmoidf_d(v);
        C[(size_t)(m0 + (wm << 5) + (mt << 4) + rAdd + r) * N + col] = v;
      }
    }
  }
}

// ---------------------------------------------------------------------------
// Sequential GRU recurrence, one block per direction, h kept in LDS,
// w_hh streamed from L2 (3 MB/step << 192 MB L2).
// ---------------------------------------------------------------------------
__global__ __launch_bounds__(IN)
void gru_seq(const float* __restrict__ gi_f, const float* __restrict__ gi_b,
             const float* __restrict__ whh_f, const float* __restrict__ whh_b,
             const float* __restrict__ bhh_f, const float* __restrict__ bhh_b,
             float* __restrict__ out_f, float* __restrict__ out_b) {
  const int dir = blockIdx.x;
  const float* gi  = dir ? gi_b  : gi_f;
  const float* whh = dir ? whh_b : whh_f;
  const float* bhh = dir ? bhh_b : bhh_f;
  float* out = dir ? out_b : out_f;

  const int j = threadIdx.x;
  __shared__ float hsh[IN];
  hsh[j] = 0.f;
  const float4* wr4 = (const float4*)(whh + (size_t)j * IN);
  const float4* wz4 = (const float4*)(whh + (size_t)(IN + j) * IN);
  const float4* wn4 = (const float4*)(whh + (size_t)(2 * IN + j) * IN);
  const float br = bhh[j], bz = bhh[IN + j], bn = bhh[2 * IN + j];
  __syncthreads();

  for (int s = 0; s < LSEQ; ++s) {
    const int t = dir ? (LSEQ - 1 - s) : s;
    float sr = 0.f, sz = 0.f, sn = 0.f;
    const float4* h4 = (const float4*)hsh;
    #pragma unroll 4
    for (int k = 0; k < IN / 4; ++k) {
      const float4 hv = h4[k];
      const float4 ar = wr4[k]; sr += hv.x*ar.x + hv.y*ar.y + hv.z*ar.z + hv.w*ar.w;
      const float4 az = wz4[k]; sz += hv.x*az.x + hv.y*az.y + hv.z*az.z + hv.w*az.w;
      const float4 an = wn4[k]; sn += hv.x*an.x + hv.y*an.y + hv.z*an.z + hv.w*an.w;
    }
    const float* girow = gi + (size_t)t * G3;   // gi already contains b_ih
    const float r = sigmoidf_d(girow[j] + sr + br);
    const float z = sigmoidf_d(girow[IN + j] + sz + bz);
    const float n = tanhf(girow[2 * IN + j] + r * (sn + bn));
    const float hprev = hsh[j];
    const float hnew = (1.f - z) * n + z * hprev;
    __syncthreads();
    hsh[j] = hnew;
    out[(size_t)t * IN + j] = hnew;
    __syncthreads();
  }
}

// ---------------------------------------------------------------------------
// Build augmented bf16 A = [silu(x) | bsplines(x)] row-major, width 9*inW.
// Cox-de Boor order-3, uniform grid over [-1,1], GS=5, K=3 -> 8 bases.
// ---------------------------------------------------------------------------
__global__ __launch_bounds__(256)
void build_aug(const float* __restrict__ src0, const float* __restrict__ src1,
               int w0, int inW, __bf16* __restrict__ Aaug) {
  const int idx = blockIdx.x * blockDim.x + threadIdx.x;
  if (idx >= LSEQ * inW) return;
  const int t = idx / inW, i = idx % inW;
  const float x = (i < w0) ? src0[(size_t)t * w0 + i]
                           : src1[(size_t)t * w0 + (i - w0)];
  const int rowW = inW * 9;
  __bf16* row = Aaug + (size_t)t * rowW;
  row[i] = f2bf(x * sigmoidf_d(x));   // silu part

  float g[12];
  #pragma unroll
  for (int jj = 0; jj < 12; ++jj) g[jj] = (float)(jj - 3) * GRID_H - 1.f;
  float b[11];
  #pragma unroll
  for (int jj = 0; jj < 11; ++jj) b[jj] = (x >= g[jj] && x < g[jj + 1]) ? 1.f : 0.f;
  #pragma unroll
  for (int k = 1; k <= 3; ++k) {
    const float inv = 1.f / ((float)k * GRID_H);
    #pragma unroll
    for (int jj = 0; jj + k < 11; ++jj)
      b[jj] = (x - g[jj]) * inv * b[jj] + (g[jj + k + 1] - x) * inv * b[jj + 1];
  }
  __bf16* sp = row + inW + (size_t)i * NB;
  #pragma unroll
  for (int gi2 = 0; gi2 < NB; ++gi2) sp[gi2] = f2bf(b[gi2]);
}

// Augmented weights: [base_w | spline_w * scaler], bf16, row width 9*inW.
__global__ __launch_bounds__(256)
void pack_waug(const float* __restrict__ base_w, const float* __restrict__ spline_w,
               const float* __restrict__ scaler, int inW, __bf16* __restrict__ Waug) {
  const int idx = blockIdx.x * blockDim.x + threadIdx.x;
  if (idx >= IN * inW) return;
  const int o = idx / inW, i = idx % inW;
  const int rowW = inW * 9;
  __bf16* row = Waug + (size_t)o * rowW;
  row[i] = f2bf(base_w[idx]);
  const float sc = scaler[idx];
  const float* sw = spline_w + (size_t)idx * NB;
  __bf16* dst = row + inW + (size_t)i * NB;
  #pragma unroll
  for (int g = 0; g < NB; ++g) dst[g] = f2bf(sw[g] * sc);
}

__global__ __launch_bounds__(256)
void cvt_bf16(const float* __restrict__ src, __bf16* __restrict__ dst, int n) {
  const int i = blockIdx.x * blockDim.x + threadIdx.x;
  if (i < n) dst[i] = f2bf(src[i]);
}

// y[o] = sum_t a2[t][o] * p[t]
__global__ __launch_bounds__(256)
void reduce_y(const float* __restrict__ a2, const float* __restrict__ p,
              float* __restrict__ y) {
  const int o = blockIdx.x * blockDim.x + threadIdx.x;
  if (o >= IN) return;
  float s = 0.f;
  for (int t = 0; t < LSEQ; ++t) s += a2[(size_t)t * IN + o] * p[t];
  y[o] = s;
}

// ---------------------------------------------------------------------------
extern "C" void kernel_launch(void* const* d_in, const int* in_sizes, int n_in,
                              void* d_out, int out_size, void* d_ws, size_t ws_size,
                              hipStream_t stream) {
  (void)in_sizes; (void)n_in; (void)out_size; (void)ws_size;
  const float* h    = (const float*)d_in[0];
  const float* p    = (const float*)d_in[1];
  const float* wihf = (const float*)d_in[2];
  const float* whhf = (const float*)d_in[3];
  const float* bihf = (const float*)d_in[4];
  const float* bhhf = (const float*)d_in[5];
  const float* wihb = (const float*)d_in[6];
  const float* whhb = (const float*)d_in[7];
  const float* bihb = (const float*)d_in[8];
  const float* bhhb = (const float*)d_in[9];
  const float* bw1  = (const float*)d_in[10];
  const float* sw1  = (const float*)d_in[11];
  const float* sc1  = (const float*)d_in[12];
  const float* bw2  = (const float*)d_in[13];
  const float* sw2  = (const float*)d_in[14];
  const float* sc2  = (const float*)d_in[15];

  float* y  = (float*)d_out;          // (512,)
  float* a1 = (float*)d_out + IN;     // (L, 512) second tuple output

  // ---- workspace layout (bytes), with aliasing of dead buffers ----
  constexpr size_t SZ_GI    = (size_t)LSEQ * G3 * 4;         // 25.2 MB each
  constexpr size_t SZ_HB    = (size_t)LSEQ * IN * 2;
  constexpr size_t SZ_WIH   = (size_t)G3 * IN * 2;
  constexpr size_t SZ_OUT   = (size_t)LSEQ * IN * 4;
  constexpr size_t SZ_W1    = (size_t)IN * IN2 * 9 * 2;
  constexpr size_t SZ_W2    = (size_t)IN * IN * 9 * 2;
  constexpr size_t OFF_GI_F  = 0;
  constexpr size_t OFF_GI_B  = OFF_GI_F + SZ_GI;
  constexpr size_t OFF_HB    = OFF_GI_B + SZ_GI;
  constexpr size_t OFF_WIHF  = OFF_HB + SZ_HB;
  constexpr size_t OFF_WIHB  = OFF_WIHF + SZ_WIH;
  constexpr size_t OFF_OUTF  = OFF_WIHB + SZ_WIH;
  constexpr size_t OFF_OUTB  = OFF_OUTF + SZ_OUT;
  constexpr size_t OFF_W1    = OFF_OUTB + SZ_OUT;
  constexpr size_t OFF_W2    = OFF_W1 + SZ_W1;
  constexpr size_t OFF_A1AUG = OFF_W2 + SZ_W2;               // 75.5 MB
  constexpr size_t OFF_A2AUG = OFF_GI_F;                     // aliases gi (dead after GRU)
  constexpr size_t OFF_A2    = OFF_A1AUG;                    // aliases A1aug (dead after KAN1)

  char* ws = (char*)d_ws;
  float*  gi_f   = (float*)(ws + OFF_GI_F);
  float*  gi_b   = (float*)(ws + OFF_GI_B);
  __bf16* hb     = (__bf16*)(ws + OFF_HB);
  __bf16* wihf_b = (__bf16*)(ws + OFF_WIHF);
  __bf16* wihb_b = (__bf16*)(ws + OFF_WIHB);
  float*  out_f  = (float*)(ws + OFF_OUTF);
  float*  out_b  = (float*)(ws + OFF_OUTB);
  __bf16* w1aug  = (__bf16*)(ws + OFF_W1);
  __bf16* w2aug  = (__bf16*)(ws + OFF_W2);
  __bf16* a1aug  = (__bf16*)(ws + OFF_A1AUG);
  __bf16* a2aug  = (__bf16*)(ws + OFF_A2AUG);
  float*  a2     = (float*)(ws + OFF_A2);

  auto blocks = [](int n) { return (n + 255) / 256; };
  auto gemm_blocks = [](int M, int N) { return (M >> 6) * (N >> 8); };

  // 1. pack bf16 operands
  cvt_bf16<<<blocks(LSEQ * IN), 256, 0, stream>>>(h, hb, LSEQ * IN);
  cvt_bf16<<<blocks(G3 * IN), 256, 0, stream>>>(wihf, wihf_b, G3 * IN);
  cvt_bf16<<<blocks(G3 * IN), 256, 0, stream>>>(wihb, wihb_b, G3 * IN);
  pack_waug<<<blocks(IN * IN2), 256, 0, stream>>>(bw1, sw1, sc1, IN2, w1aug);
  pack_waug<<<blocks(IN * IN), 256, 0, stream>>>(bw2, sw2, sc2, IN, w2aug);

  // 2. gi = h @ w_ih^T + b_ih (both directions), WMMA bf16
  wmma_gemm<<<gemm_blocks(LSEQ, G3), 256, 0, stream>>>(hb, wihf_b, bihf, gi_f, LSEQ, G3, IN, 0);
  wmma_gemm<<<gemm_blocks(LSEQ, G3), 256, 0, stream>>>(hb, wihb_b, bihb, gi_b, LSEQ, G3, IN, 0);

  // 3. sequential recurrence (one block per direction)
  gru_seq<<<2, IN, 0, stream>>>(gi_f, gi_b, whhf, whhb, bhhf, bhhb, out_f, out_b);

  // 4. KAN1: augmented A from concat(out_f,out_b), GEMM K=9216, ReLU -> a1 (into d_out)
  build_aug<<<blocks(LSEQ * IN2), 256, 0, stream>>>(out_f, out_b, IN, IN2, a1aug);
  wmma_gemm<<<gemm_blocks(LSEQ, IN), 256, 0, stream>>>(a1aug, w1aug, nullptr, a1,
                                                      LSEQ, IN, IN2 * 9, 1);

  // 5. KAN2: augmented A from a1, GEMM K=4608, sigmoid -> a2
  build_aug<<<blocks(LSEQ * IN), 256, 0, stream>>>(a1, a1, IN, IN, a2aug);
  wmma_gemm<<<gemm_blocks(LSEQ, IN), 256, 0, stream>>>(a2aug, w2aug, nullptr, a2,
                                                      LSEQ, IN, IN * 9, 2);

  // 6. y = a2^T @ p
  reduce_y<<<2, 256, 0, stream>>>(a2, p, y);
}
